// GroupedQueryAttention_20641612825240
// MI455X (gfx1250) — compile-verified, matrix operand order checked by reading, not run on hardware
//
#include <hip/hip_runtime.h>

#define S_LEN   2048
#define DMODEL  4096
#define NQ      32
#define NKV     8
#define HD_     128
#define WINDOW_ 1024

typedef __attribute__((ext_vector_type(16))) __bf16 v16bf;
typedef __attribute__((ext_vector_type(8)))  float  v8f;

union Frag16 { uint4 u[2]; v16bf v; };

__device__ __forceinline__ unsigned short f2bf(float f) {
  __bf16 h = (__bf16)f;
  return __builtin_bit_cast(unsigned short, h);
}

__device__ __forceinline__ unsigned int pack2bf(float a, float b) {
  return (unsigned int)f2bf(a) | ((unsigned int)f2bf(b) << 16);
}

__device__ __forceinline__ v8f wmma_bf16(v16bf a, v16bf b, v8f c) {
  // D = A(16x32 bf16) * B(32x16 bf16) + C(16x16 f32)
  return __builtin_amdgcn_wmma_f32_16x16x32_bf16(false, a, false, b, (short)0, c,
                                                 false, false);
}

// ---------------------------------------------------------------------------
// Tiled GEMM: C[MxN] = A[MxK] * B[KxN], f32 in/out, bf16 WMMA compute.
// N, K compile-time so row strides fold into store/load offset immediates.
// Block = 256 threads (8 waves), tile 64(M) x 128(N), K-step 32.
// Double-buffered LDS ping-pong, manually 2x unrolled so the buffer index is
// a compile-time constant; one barrier per K-step.
// ---------------------------------------------------------------------------
template <int N, int K>
__global__ __launch_bounds__(256)
void gemm_bf16_wmma(const float* __restrict__ A, const float* __restrict__ B,
                    float* __restrict__ C) {
  __shared__ __align__(16) unsigned short As[2][64][32];   // row-major (2x4 KB)
  __shared__ __align__(16) unsigned short Bs[2][128][32];  // transposed [n][k] (2x8 KB)

  const int tid  = threadIdx.x;
  const int lane = tid & 31;
  const int wid  = tid >> 5;
  const int ln   = lane & 15;
  const int hi   = lane >> 4;
  const int bm   = blockIdx.y * 64;
  const int bn   = blockIdx.x * 128;
  const int waveM = (wid & 1) * 32;
  const int waveN = (wid >> 1) * 32;

  // staging assignments
  const int arow = tid >> 2;          // 0..63
  const int acg  = (tid & 3) * 8;     // 0,8,16,24
  const int bn0  = (tid & 15) * 8;    // 0..120 step 8
  const int bk2  = (tid >> 4) * 2;    // 0..30 step 2

  union { unsigned short us[8]; uint4 u; } aStage;
  unsigned int bStage[8];

  const float* aBase = A + (size_t)(bm + arow) * K + acg;
  const float* bBase = B + (size_t)bk2 * N + bn + bn0;

  auto stage = [&](int k0) {
    // A chunk: 8 consecutive k of one row
    const float* sa = aBase + k0;
    const float4 a0 = ((const float4*)sa)[0];
    const float4 a1 = ((const float4*)sa)[1];
    aStage.us[0] = f2bf(a0.x); aStage.us[1] = f2bf(a0.y);
    aStage.us[2] = f2bf(a0.z); aStage.us[3] = f2bf(a0.w);
    aStage.us[4] = f2bf(a1.x); aStage.us[5] = f2bf(a1.y);
    aStage.us[6] = f2bf(a1.z); aStage.us[7] = f2bf(a1.w);
    // B chunk: 2 consecutive k rows x 8 n, packed pairwise along k
    const float* s0 = bBase + (size_t)k0 * N;
    const float* s1 = s0 + N;
    const float4 p00 = ((const float4*)s0)[0];
    const float4 p01 = ((const float4*)s0)[1];
    const float4 p10 = ((const float4*)s1)[0];
    const float4 p11 = ((const float4*)s1)[1];
    bStage[0] = pack2bf(p00.x, p10.x); bStage[1] = pack2bf(p00.y, p10.y);
    bStage[2] = pack2bf(p00.z, p10.z); bStage[3] = pack2bf(p00.w, p10.w);
    bStage[4] = pack2bf(p01.x, p11.x); bStage[5] = pack2bf(p01.y, p11.y);
    bStage[6] = pack2bf(p01.z, p11.z); bStage[7] = pack2bf(p01.w, p11.w);
  };

  v8f acc[2][2] = {};

  // one K-step against buffer `buf` (compile-time), optionally staging k0+32
  auto step = [&](int buf, int k0, bool more) {
    if (more) stage(k0 + 32);  // global loads overlap the WMMAs below

    Frag16 af[2], bf[2];
    #pragma unroll
    for (int s = 0; s < 2; ++s) {
      const int r = waveM + s * 16 + ln;
      af[s].u[0] = *(const uint4*)&As[buf][r][hi * 8];
      af[s].u[1] = *(const uint4*)&As[buf][r][16 + hi * 8];
    }
    #pragma unroll
    for (int s = 0; s < 2; ++s) {
      const int n = waveN + s * 16 + ln;
      bf[s].u[0] = *(const uint4*)&Bs[buf][n][hi * 8];
      bf[s].u[1] = *(const uint4*)&Bs[buf][n][16 + hi * 8];
    }
    #pragma unroll
    for (int i = 0; i < 2; ++i)
      #pragma unroll
      for (int j = 0; j < 2; ++j)
        acc[i][j] = wmma_bf16(af[i].v, bf[j].v, acc[i][j]);

    if (more) {  // commit to the other buffer: no conflict with readers of buf
      const int nb = buf ^ 1;
      *(uint4*)&As[nb][arow][acg] = aStage.u;
      #pragma unroll
      for (int i = 0; i < 8; ++i)
        *(unsigned int*)&Bs[nb][bn0 + i][bk2] = bStage[i];
    }
    __syncthreads();
  };

  stage(0);
  *(uint4*)&As[0][arow][acg] = aStage.u;
  #pragma unroll
  for (int i = 0; i < 8; ++i)
    *(unsigned int*)&Bs[0][bn0 + i][bk2] = bStage[i];
  __syncthreads();

  // K/32 is even for all instantiations: 2x-unrolled ping-pong, no tail
  for (int k0 = 0; k0 < K; k0 += 64) {
    step(0, k0, true);
    step(1, k0 + 32, (k0 + 64) < K);
  }

  // ---- epilogue: base pointer per sub-tile, rows fold into offset imms ----
  #pragma unroll
  for (int i = 0; i < 2; ++i)
    #pragma unroll
    for (int j = 0; j < 2; ++j) {
      const int row0 = bm + waveM + i * 16 + 8 * hi;
      const int col  = bn + waveN + j * 16 + ln;
      float* p = C + (size_t)row0 * N + col;
      #pragma unroll
      for (int r = 0; r < 8; ++r)
        p[(size_t)r * N] = acc[i][j][r];   // N constant => offset immediate
    }
}

// ---------------------------------------------------------------------------
// RoPE + bf16 pack for Q: xq f32 [s][h*128+d] -> Qp bf16 [h][s][d], * 1/sqrt(HD)
// ---------------------------------------------------------------------------
__global__ void rope_pack_q(const float* __restrict__ xq, const float* __restrict__ fc,
                            const float* __restrict__ fs, unsigned short* __restrict__ Qp) {
  const int idx = blockIdx.x * blockDim.x + threadIdx.x;  // S*NQ*64 pairs
  const int p = idx & 63;
  const int h = (idx >> 6) & 31;
  const int s = idx >> 11;
  const float e = xq[(size_t)s * DMODEL + h * HD_ + 2 * p];
  const float o = xq[(size_t)s * DMODEL + h * HD_ + 2 * p + 1];
  const float c = fc[s * 64 + p], sn = fs[s * 64 + p];
  const float scale = 0.08838834764831845f;  // 1/sqrt(128), folded into Q
  *(unsigned int*)&Qp[((size_t)h * S_LEN + s) * HD_ + 2 * p] =
      pack2bf((e * c - o * sn) * scale, (e * sn + o * c) * scale);
}

// RoPE + pack K: xk f32 [s][h*128+d] -> Kp bf16 [h][s][d]
__global__ void rope_pack_k(const float* __restrict__ xk, const float* __restrict__ fc,
                            const float* __restrict__ fs, unsigned short* __restrict__ Kp) {
  const int idx = blockIdx.x * blockDim.x + threadIdx.x;  // S*NKV*64 pairs
  const int p = idx & 63;
  const int h = (idx >> 6) & 7;
  const int s = idx >> 9;
  const float e = xk[(size_t)s * (NKV * HD_) + h * HD_ + 2 * p];
  const float o = xk[(size_t)s * (NKV * HD_) + h * HD_ + 2 * p + 1];
  const float c = fc[s * 64 + p], sn = fs[s * 64 + p];
  *(unsigned int*)&Kp[((size_t)h * S_LEN + s) * HD_ + 2 * p] =
      pack2bf(e * c - o * sn, e * sn + o * c);
}

// Pack V transposed: xv f32 [s][h*128+d] -> Vp bf16 [h][d][s]
__global__ void pack_v(const float* __restrict__ xv, unsigned short* __restrict__ Vp) {
  const int idx = blockIdx.x * blockDim.x + threadIdx.x;  // S*NKV*HD
  const int d = idx & 127;
  const int h = (idx >> 7) & 7;
  const int s = idx >> 10;
  Vp[((size_t)h * HD_ + d) * S_LEN + s] = f2bf(xv[(size_t)s * (NKV * HD_) + h * HD_ + d]);
}

// ---------------------------------------------------------------------------
// Flash attention: 1 wave per (head, 16-row q tile). Sliding-window causal.
// ---------------------------------------------------------------------------
__global__ __launch_bounds__(32)
void attn_kernel(const unsigned short* __restrict__ Qp,  // [NQ][S][HD] bf16
                 const unsigned short* __restrict__ Kp,  // [NKV][S][HD] bf16
                 const unsigned short* __restrict__ Vp,  // [NKV][HD][S] bf16
                 float* __restrict__ Oout) {             // [S][NQ*HD] f32
  __shared__ __align__(16) unsigned short Pl[16][32];    // P tile staging (1 KB)

  const int lane  = threadIdx.x;
  const int ln    = lane & 15;
  const int hi    = lane >> 4;
  const int h     = blockIdx.x;        // query head
  const int qbase = blockIdx.y * 16;
  const int kvh   = h >> 2;            // N_REP = 4

  // Q fragments resident: A-layout, lane -> row = qbase+ln, slots along d
  const unsigned short* qrow = Qp + ((size_t)h * S_LEN + (qbase + ln)) * HD_;
  Frag16 qf[4];
  #pragma unroll
  for (int f = 0; f < 4; ++f) {
    qf[f].u[0] = *(const uint4*)(qrow + f * 32 + hi * 8);
    qf[f].u[1] = *(const uint4*)(qrow + f * 32 + 16 + hi * 8);
  }

  v8f oacc[8] = {};
  float mrow[8], lrow[8];
  #pragma unroll
  for (int j = 0; j < 8; ++j) { mrow[j] = -3.0e38f; lrow[j] = 0.0f; }

  int lo = qbase - (WINDOW_ - 1);
  if (lo < 0) lo = 0;
  lo &= ~31;
  const int hiKey = qbase + 15;

  for (int kb = lo; kb <= hiKey; kb += 32) {
    // ---- scores: two 16x16 C tiles covering keys kb..kb+31 ----
    v8f sc[2] = {};
    #pragma unroll
    for (int t = 0; t < 2; ++t) {
      const unsigned short* krow = Kp + ((size_t)kvh * S_LEN + (kb + t * 16 + ln)) * HD_;
      #pragma unroll
      for (int f = 0; f < 4; ++f) {
        Frag16 kf;
        kf.u[0] = *(const uint4*)(krow + f * 32 + hi * 8);
        kf.u[1] = *(const uint4*)(krow + f * 32 + 16 + hi * 8);
        sc[t] = wmma_bf16(qf[f].v, kf.v, sc[t]);
      }
    }

    // ---- mask (causal + sliding window, computed analytically) ----
    float rmax[8];
    #pragma unroll
    for (int j = 0; j < 8; ++j) rmax[j] = -3.0e38f;
    #pragma unroll
    for (int t = 0; t < 2; ++t)
      #pragma unroll
      for (int j = 0; j < 8; ++j) {
        const int qg  = qbase + j + 8 * hi;
        const int key = kb + t * 16 + ln;
        const int dfs = qg - key;
        const bool valid = (dfs >= 0) && (dfs < WINDOW_);
        const float s = valid ? sc[t][j] : -3.0e38f;
        sc[t][j] = s;
        rmax[j] = fmaxf(rmax[j], s);
      }
    // row-max across the 16-lane half-group (rows live in one VGPR)
    #pragma unroll
    for (int off = 8; off >= 1; off >>= 1)
      #pragma unroll
      for (int j = 0; j < 8; ++j)
        rmax[j] = fmaxf(rmax[j], __shfl_xor(rmax[j], off, 32));

    float alpha[8];
    #pragma unroll
    for (int j = 0; j < 8; ++j) {
      const float mnew = fmaxf(mrow[j], rmax[j]);
      alpha[j] = __expf(mrow[j] - mnew);
      mrow[j] = mnew;
    }

    // ---- probs -> LDS (C-layout -> A-layout via shared) + row sums ----
    float rsum[8];
    #pragma unroll
    for (int j = 0; j < 8; ++j) rsum[j] = 0.0f;
    #pragma unroll
    for (int t = 0; t < 2; ++t)
      #pragma unroll
      for (int j = 0; j < 8; ++j) {
        float p = __expf(sc[t][j] - mrow[j]);
        if (sc[t][j] <= -1.0e37f) p = 0.0f;  // masked
        rsum[j] += p;
        Pl[j + 8 * hi][t * 16 + ln] = f2bf(p);
      }
    #pragma unroll
    for (int off = 8; off >= 1; off >>= 1)
      #pragma unroll
      for (int j = 0; j < 8; ++j) rsum[j] += __shfl_xor(rsum[j], off, 32);
    #pragma unroll
    for (int j = 0; j < 8; ++j) lrow[j] = lrow[j] * alpha[j] + rsum[j];

    asm volatile("s_wait_dscnt 0" ::: "memory");  // single-wave WG: DS is in-order

    Frag16 pf;
    pf.u[0] = *(const uint4*)&Pl[ln][hi * 8];
    pf.u[1] = *(const uint4*)&Pl[ln][16 + hi * 8];

    // ---- O = O*alpha + P @ V  (V transposed pack gives contiguous B frags) ----
    #pragma unroll
    for (int g = 0; g < 8; ++g) {
      #pragma unroll
      for (int j = 0; j < 8; ++j) oacc[g][j] *= alpha[j];
      const unsigned short* vcol = Vp + ((size_t)kvh * HD_ + g * 16 + ln) * S_LEN + kb;
      Frag16 vf;
      vf.u[0] = *(const uint4*)(vcol + hi * 8);
      vf.u[1] = *(const uint4*)(vcol + 16 + hi * 8);
      oacc[g] = wmma_bf16(pf.v, vf.v, oacc[g]);
    }
  }

  // ---- epilogue: normalize; rows fold into offset-immediate stores ----
  #pragma unroll
  for (int g = 0; g < 8; ++g) {
    float* p = Oout + (size_t)(qbase + 8 * hi) * DMODEL + h * HD_ + g * 16 + ln;
    #pragma unroll
    for (int j = 0; j < 8; ++j)
      p[(size_t)j * DMODEL] = oacc[g][j] / lrow[j];
  }
}

// ---------------------------------------------------------------------------
extern "C" void kernel_launch(void* const* d_in, const int* in_sizes, int n_in,
                              void* d_out, int out_size, void* d_ws, size_t ws_size,
                              hipStream_t stream) {
  (void)in_sizes; (void)n_in; (void)out_size; (void)ws_size;

  const float* x  = (const float*)d_in[0];
  const float* fc = (const float*)d_in[1];
  const float* fs = (const float*)d_in[2];
  /* d_in[3] = mask: recomputed analytically in-kernel */
  const float* wq = (const float*)d_in[4];
  const float* wk = (const float*)d_in[5];
  const float* wv = (const float*)d_in[6];
  const float* wo = (const float*)d_in[7];
  float* out = (float*)d_out;

  char* ws = (char*)d_ws;
  float*          xq   = (float*)(ws + 0);                    // 33.5 MB
  float*          xk   = (float*)(ws + 33554432);             //  8.4 MB
  float*          xv   = (float*)(ws + 41943040);             //  8.4 MB
  float*          attn = (float*)(ws + 50331648);             // 33.5 MB
  unsigned short* Qp   = (unsigned short*)(ws + 83886080);    // 16.8 MB
  unsigned short* Kp   = (unsigned short*)(ws + 100663296);   //  4.2 MB
  unsigned short* Vp   = (unsigned short*)(ws + 104857600);   //  4.2 MB

  // QKV projections
  gemm_bf16_wmma<DMODEL, DMODEL><<<dim3(DMODEL / 128, S_LEN / 64), 256, 0, stream>>>(x, wq, xq);
  gemm_bf16_wmma<NKV * HD_, DMODEL><<<dim3((NKV * HD_) / 128, S_LEN / 64), 256, 0, stream>>>(x, wk, xk);
  gemm_bf16_wmma<NKV * HD_, DMODEL><<<dim3((NKV * HD_) / 128, S_LEN / 64), 256, 0, stream>>>(x, wv, xv);

  // RoPE + bf16 packing
  rope_pack_q<<<(S_LEN * NQ * 64) / 256, 256, 0, stream>>>(xq, fc, fs, Qp);
  rope_pack_k<<<(S_LEN * NKV * 64) / 256, 256, 0, stream>>>(xk, fc, fs, Kp);
  pack_v<<<(S_LEN * NKV * HD_) / 256, 256, 0, stream>>>(xv, Vp);

  // Flash attention (32 heads x 128 q-tiles)
  attn_kernel<<<dim3(NQ, S_LEN / 16), 32, 0, stream>>>(Qp, Kp, Vp, attn);

  // Output projection
  gemm_bf16_wmma<DMODEL, DMODEL><<<dim3(DMODEL / 128, S_LEN / 64), 256, 0, stream>>>(attn, wo, out);
}
